// Transformer_74895639708189
// MI455X (gfx1250) — matrix operand order repaired and checked by reading, not compiled, not run on hardware
//
#include <hip/hip_runtime.h>
#include <math.h>

// ---------------- model constants ----------------
#define DMODEL 256
#define NHEAD  8
#define HDIM   32
#define NLVL   4
#define NPTS   4
#define DFF    1024
#define NLAYERS 6
#define BSZ    2
#define SEQ    13294   // 100*100 + 50*50 + 25*25 + 13*13

// ---------------- vector types ----------------
typedef __attribute__((ext_vector_type(16))) __bf16 v16bf;
typedef __attribute__((ext_vector_type(8)))  __bf16 v8bf;
typedef __attribute__((ext_vector_type(4)))  __bf16 v4bf;
typedef __attribute__((ext_vector_type(8)))  float  v8f;
typedef __attribute__((ext_vector_type(4)))  unsigned u32x4;
typedef __attribute__((ext_vector_type(8)))  unsigned u32x8;

#define TSTR 40  // LDS row stride in bf16 elems (64B row + 16B pad = 80B)

// =====================================================================
// GEMM: C[M,N] = A[M,K] @ W[K,N] + bias   (A, W bf16; C fp32 and/or bf16)
// Block = 128x128 tile, 8 waves; wave = 64x32 = 4x2 WMMA tiles.
// A tile is DMA'd into LDS by the Tensor Data Mover (tensor_load_to_lds):
//   - tensor_dim1 = M-blockM -> OOB rows read as zero (free M-edge handling)
//   - pad_enable: +16B after every 64B row -> lands in padded TSTR layout
// B tile is transposed into LDS [n][k] via 4x4 blocks (packed b64 stores).
// =====================================================================
template <bool RELU, bool OUTF32, bool OUTB16>
__global__ __launch_bounds__(256)
void gemm_bias_kernel(const __bf16* __restrict__ A, const __bf16* __restrict__ W,
                      const float* __restrict__ bias,
                      float* __restrict__ C, __bf16* __restrict__ Cb,
                      int M, int K, int N)
{
    __shared__ __bf16 As[128 * TSTR];   // [m][k] 128x32, padded, filled by TDM
    __shared__ __bf16 Bs[128 * TSTR];   // [n][k] 128x32 transposed, padded

    const int t    = threadIdx.x;
    const int lane = t & 31;
    const int wid  = t >> 5;
    const int waveM = (wid >> 2) * 64;
    const int waveN = (wid & 3) * 32;
    const int half  = lane >> 4;
    const int l16   = lane & 15;

    const int blockM = blockIdx.y * 128;
    const int blockN = blockIdx.x * 128;

    // B transpose assignment: each thread owns a 4(k) x 4(n) block
    const int kb = (t & 7) * 4;          // 0..28
    const int nb = (t >> 3) * 4;         // 0..124

    const unsigned lds_off = (unsigned)(size_t)&As[0];  // low 32b = LDS offset

    v8f acc[4][2];
    #pragma unroll
    for (int i = 0; i < 4; ++i)
        #pragma unroll
        for (int j = 0; j < 2; ++j)
            #pragma unroll
            for (int r = 0; r < 8; ++r) acc[i][j][r] = 0.0f;

    const int KT = K >> 5;
    for (int kt = 0; kt < KT; ++kt) {
        // ---- A tile via Tensor Data Mover (wave 0 issues, all waves barrier)
        if (wid == 0) {
            const unsigned long long ga =
                (unsigned long long)(size_t)(A + (size_t)blockM * K + (size_t)kt * 32);
            const unsigned rows_rem = (unsigned)(M - blockM);
            u32x4 g0;
            g0[0] = 1u;                                   // count=1, user mode
            g0[1] = lds_off;                              // lds_addr
            g0[2] = (unsigned)ga;                         // global_addr[31:0]
            g0[3] = (unsigned)((ga >> 32) & 0x1FFFFFFu)   // global_addr[56:32]
                  | (2u << 30);                           // type = 2 (image)
            u32x8 g1;
            g1[0] = (1u << 16)        // data_size = 1 (2 bytes)
                  | (1u << 20)        // pad_enable
                  | (3u << 22)        // pad_interval: every 16 DWORDs (64B)
                  | (3u << 25);       // pad_amount: 4 DWORDs (16B)
            g1[1] = (32u << 16);                          // tensor_dim0[15:0]=32
            g1[2] = (rows_rem & 0xFFFFu) << 16;           // tensor_dim1[15:0]
            g1[3] = (rows_rem >> 16) | (32u << 16);       // dim1[31:16] | tile_dim0=32
            g1[4] = 128u;                                 // tile_dim1=128, tile_dim2=0
            g1[5] = (unsigned)K;                          // tensor_dim0_stride
            g1[6] = 0u;
            g1[7] = 0u;
            u32x4 z4; z4[0] = 0u; z4[1] = 0u; z4[2] = 0u; z4[3] = 0u;
            asm volatile("tensor_load_to_lds %0, %1, %2, %3"
                         :: "s"(g0), "s"(g1), "s"(z4), "s"(z4) : "memory");
            __builtin_amdgcn_s_wait_tensorcnt(0);
        }

        // ---- B tile: 32x128 bf16 -> LDS transposed [n][k], packed b64 stores
        v4bf r0 = *(const v4bf*)(W + (size_t)(kt * 32 + kb + 0) * N + blockN + nb);
        v4bf r1 = *(const v4bf*)(W + (size_t)(kt * 32 + kb + 1) * N + blockN + nb);
        v4bf r2 = *(const v4bf*)(W + (size_t)(kt * 32 + kb + 2) * N + blockN + nb);
        v4bf r3 = *(const v4bf*)(W + (size_t)(kt * 32 + kb + 3) * N + blockN + nb);
        if (kt + 1 < KT)
            __builtin_prefetch(W + (size_t)((kt + 1) * 32 + kb) * N + blockN + nb, 0, 0);
        #pragma unroll
        for (int n = 0; n < 4; ++n) {
            v4bf col; col[0] = r0[n]; col[1] = r1[n]; col[2] = r2[n]; col[3] = r3[n];
            *(v4bf*)&Bs[(nb + n) * TSTR + kb] = col;
        }
        __syncthreads();

        // ---- fragment assembly (ISA 16-bit A 16x32 / B 32x16 layouts) ----
        v16bf afrag[4];
        #pragma unroll
        for (int mi = 0; mi < 4; ++mi) {
            const __bf16* ap = &As[(waveM + mi * 16 + l16) * TSTR + half * 8];
            v8bf lo = *(const v8bf*)ap;
            v8bf hi = *(const v8bf*)(ap + 16);
            v16bf a;
            #pragma unroll
            for (int e = 0; e < 8; ++e) { a[e] = lo[e]; a[e + 8] = hi[e]; }
            afrag[mi] = a;
        }
        v16bf bfrag[2];
        #pragma unroll
        for (int ni = 0; ni < 2; ++ni) {
            const __bf16* bp = &Bs[(waveN + ni * 16 + l16) * TSTR + half * 8];
            v8bf lo = *(const v8bf*)bp;
            v8bf hi = *(const v8bf*)(bp + 16);
            v16bf b;
            #pragma unroll
            for (int e = 0; e < 8; ++e) { b[e] = lo[e]; b[e + 8] = hi[e]; }
            bfrag[ni] = b;
        }

        #pragma unroll
        for (int mi = 0; mi < 4; ++mi)
            #pragma unroll
            for (int ni = 0; ni < 2; ++ni)
                acc[mi][ni] = __builtin_amdgcn_wmma_f32_16x16x32_bf16(
                    false, afrag[mi], false, bfrag[ni],
                    (short)0, acc[mi][ni], false, false);
        __syncthreads();
    }

    // ---- epilogue ----
    #pragma unroll
    for (int mi = 0; mi < 4; ++mi) {
        #pragma unroll
        for (int ni = 0; ni < 2; ++ni) {
            const int col = blockN + waveN + ni * 16 + l16;
            const float bv = bias[col];
            #pragma unroll
            for (int r = 0; r < 8; ++r) {
                const int row = blockM + waveM + mi * 16 + r + half * 8;
                if (row < M) {
                    float o = acc[mi][ni][r] + bv;
                    if (RELU) o = fmaxf(o, 0.0f);
                    if (OUTF32) C [(size_t)row * N + col] = o;
                    if (OUTB16) Cb[(size_t)row * N + col] = (__bf16)o;
                }
            }
        }
    }
}

// =====================================================================
// fp32 -> bf16 conversion (float4 in, b64 out)
// =====================================================================
__global__ __launch_bounds__(256)
void cvt_bf16_kernel(const float* __restrict__ in, __bf16* __restrict__ out, int n4)
{
    int i = blockIdx.x * blockDim.x + threadIdx.x;
    if (i < n4) {
        float4 v = ((const float4*)in)[i];
        v4bf o; o[0] = (__bf16)v.x; o[1] = (__bf16)v.y; o[2] = (__bf16)v.z; o[3] = (__bf16)v.w;
        ((v4bf*)out)[i] = o;
    }
}

// =====================================================================
// q = bf16(src + pos)
// =====================================================================
__global__ __launch_bounds__(256)
void addq_bf16_kernel(const float4* __restrict__ a, const float4* __restrict__ b,
                      __bf16* __restrict__ o, int n4)
{
    int i = blockIdx.x * blockDim.x + threadIdx.x;
    if (i < n4) {
        float4 x = a[i], y = b[i];
        v4bf q; q[0] = (__bf16)(x.x + y.x); q[1] = (__bf16)(x.y + y.y);
                q[2] = (__bf16)(x.z + y.z); q[3] = (__bf16)(x.w + y.w);
        ((v4bf*)o)[i] = q;
    }
}

// =====================================================================
// reference points
// =====================================================================
__global__ __launch_bounds__(256)
void refpoints_kernel(const float* __restrict__ vr, float* __restrict__ ref)
{
    int i = blockIdx.x * blockDim.x + threadIdx.x;
    if (i >= BSZ * SEQ) return;
    const int b = i / SEQ, s = i % SEQ;
    int lvl, q;
    if      (s < 10000) { lvl = 0; q = s; }
    else if (s < 12500) { lvl = 1; q = s - 10000; }
    else if (s < 13125) { lvl = 2; q = s - 12500; }
    else                { lvl = 3; q = s - 13125; }
    const int HW[4] = {100, 50, 25, 13};
    const int Hh = HW[lvl], Ww = HW[lvl];
    const int iy = q / Ww, ix = q % Ww;
    const float vx = vr[(b * NLVL + lvl) * 2 + 0];
    const float vy = vr[(b * NLVL + lvl) * 2 + 1];
    const float rx = (ix + 0.5f) / (vx * (float)Ww);
    const float ry = (iy + 0.5f) / (vy * (float)Hh);
    #pragma unroll
    for (int l2 = 0; l2 < NLVL; ++l2) {
        const float v2x = vr[(b * NLVL + l2) * 2 + 0];
        const float v2y = vr[(b * NLVL + l2) * 2 + 1];
        ref[(((size_t)b * SEQ + s) * NLVL + l2) * 2 + 0] = rx * v2x;
        ref[(((size_t)b * SEQ + s) * NLVL + l2) * 2 + 1] = ry * v2y;
    }
}

// =====================================================================
// softmax over last-16 groups, in place
// =====================================================================
__global__ __launch_bounds__(256)
void softmax16_kernel(float* __restrict__ attn, int rows)
{
    int r = blockIdx.x * blockDim.x + threadIdx.x;
    if (r >= rows) return;
    float* p = attn + (size_t)r * 16;
    float mx = -3.0e38f;
    float e[16];
    #pragma unroll
    for (int i = 0; i < 16; ++i) mx = fmaxf(mx, p[i]);
    float s = 0.0f;
    #pragma unroll
    for (int i = 0; i < 16; ++i) { e[i] = __expf(p[i] - mx); s += e[i]; }
    const float inv = 1.0f / s;
    #pragma unroll
    for (int i = 0; i < 16; ++i) p[i] = e[i] * inv;
}

// =====================================================================
// Multi-scale deformable sampling; wave per (b,s,h), lane = channel.
// Gathers hit L2 (level tensors <= 10MB << 192MB). Output bf16 (feeds GEMM).
// =====================================================================
__global__ __launch_bounds__(256)
void deform_kernel(const float* __restrict__ val, const float* __restrict__ off,
                   const float* __restrict__ attn, const float* __restrict__ ref,
                   __bf16* __restrict__ out)
{
    const int lane = threadIdx.x & 31;
    const int w = blockIdx.x * (blockDim.x >> 5) + (threadIdx.x >> 5);
    const int h  = w % NHEAD;
    const int bs = w / NHEAD;
    const int HW[4]    = {100, 50, 25, 13};
    const int START[4] = {0, 10000, 12500, 13125};
    const float* offp = off  + ((size_t)bs * NHEAD + h) * (NLVL * NPTS * 2);
    const float* attp = attn + ((size_t)bs * NHEAD + h) * (NLVL * NPTS);
    const float* refp = ref  + (size_t)bs * NLVL * 2;
    const size_t vbase = (size_t)(bs / SEQ) * SEQ;

    float acc = 0.0f;
    #pragma unroll
    for (int lvl = 0; lvl < NLVL; ++lvl) {
        const int Hh = HW[lvl], Ww = HW[lvl];
        const int st = START[lvl];
        const float rx = refp[lvl * 2 + 0];
        const float ry = refp[lvl * 2 + 1];
        #pragma unroll
        for (int p = 0; p < NPTS; ++p) {
            const float ox = offp[(lvl * NPTS + p) * 2 + 0];
            const float oy = offp[(lvl * NPTS + p) * 2 + 1];
            const float x = (rx + ox / (float)Ww) * (float)Ww - 0.5f;
            const float y = (ry + oy / (float)Hh) * (float)Hh - 0.5f;
            const float x0 = floorf(x), y0 = floorf(y);
            const float wx1 = x - x0, wy1 = y - y0;
            const int ix0 = (int)x0, iy0 = (int)y0;
            float samp = 0.0f;
            #pragma unroll
            for (int dy = 0; dy < 2; ++dy) {
                #pragma unroll
                for (int dx = 0; dx < 2; ++dx) {
                    const int xi = ix0 + dx, yi = iy0 + dy;
                    const float wgt = (dx ? wx1 : 1.0f - wx1) *
                                      (dy ? wy1 : 1.0f - wy1);
                    const bool ok = (xi >= 0) & (xi < Ww) & (yi >= 0) & (yi < Hh);
                    const int xc = min(max(xi, 0), Ww - 1);
                    const int yc = min(max(yi, 0), Hh - 1);
                    const int fl = yc * Ww + xc;
                    const float gv =
                        val[((vbase + st + fl) * NHEAD + h) * HDIM + lane];
                    samp += ok ? gv * wgt : 0.0f;
                }
            }
            acc += samp * attp[lvl * NPTS + p];
        }
    }
    out[((size_t)bs * NHEAD + h) * HDIM + lane] = (__bf16)acc;
}

// =====================================================================
// out = LayerNorm(X + Y) * g + b, optional bf16 mirror (wave per row)
// =====================================================================
template <bool WB16>
__global__ __launch_bounds__(256)
void resid_ln_kernel(const float* __restrict__ X, const float* __restrict__ Y,
                     const float* __restrict__ g, const float* __restrict__ b,
                     float* __restrict__ out, __bf16* __restrict__ outb, int M)
{
    const int lane = threadIdx.x & 31;
    const int row  = blockIdx.x * 8 + (threadIdx.x >> 5);
    if (row >= M) return;
    const float* xr = X + (size_t)row * DMODEL;
    const float* yr = Y + (size_t)row * DMODEL;
    float v[8];
    float s = 0.0f, ss = 0.0f;
    #pragma unroll
    for (int i = 0; i < 8; ++i) {
        const float t = xr[lane + 32 * i] + yr[lane + 32 * i];
        v[i] = t; s += t; ss += t * t;
    }
    #pragma unroll
    for (int m = 16; m >= 1; m >>= 1) {
        s  += __shfl_xor(s,  m, 32);
        ss += __shfl_xor(ss, m, 32);
    }
    const float mean = s * (1.0f / DMODEL);
    const float var  = ss * (1.0f / DMODEL) - mean * mean;
    const float rinv = rsqrtf(var + 1e-5f);
    #pragma unroll
    for (int i = 0; i < 8; ++i) {
        const int c = lane + 32 * i;
        const float o = (v[i] - mean) * rinv * g[c] + b[c];
        out[(size_t)row * DMODEL + c] = o;
        if (WB16) outb[(size_t)row * DMODEL + c] = (__bf16)o;
    }
}

// =====================================================================
// launcher
// =====================================================================
extern "C" void kernel_launch(void* const* d_in, const int* in_sizes, int n_in,
                              void* d_out, int out_size, void* d_ws, size_t ws_size,
                              hipStream_t stream)
{
    (void)in_sizes; (void)n_in; (void)out_size; (void)ws_size;
    const float* src    = (const float*)d_in[0];
    const float* pos    = (const float*)d_in[1];
    const float* vr     = (const float*)d_in[2];
    const float* W_off  = (const float*)d_in[3];
    const float* b_off  = (const float*)d_in[4];
    const float* W_attn = (const float*)d_in[5];
    const float* b_attn = (const float*)d_in[6];
    const float* W_val  = (const float*)d_in[7];
    const float* b_val  = (const float*)d_in[8];
    const float* W_out  = (const float*)d_in[9];
    const float* b_out  = (const float*)d_in[10];
    const float* g1     = (const float*)d_in[11];
    const float* be1    = (const float*)d_in[12];
    const float* W_ff1  = (const float*)d_in[13];
    const float* b_ff1  = (const float*)d_in[14];
    const float* W_ff2  = (const float*)d_in[15];
    const float* b_ff2  = (const float*)d_in[16];
    const float* g2     = (const float*)d_in[17];
    const float* be2    = (const float*)d_in[18];

    const size_t M = (size_t)BSZ * SEQ;       // 26588 rows

    // ---- workspace carve: fp32 region then bf16 region ----
    float* ws = (float*)d_ws;
    float* valbuf  = ws; ws += M * DMODEL;    // value; reused as src2/ff2 (tmp)
    float* offbuf  = ws; ws += M * DMODEL;
    float* srcA    = ws; ws += M * DMODEL;
    float* srcB    = ws; ws += M * DMODEL;
    float* attnbuf = ws; ws += M * (NHEAD * NLVL * NPTS);
    float* refbuf  = ws; ws += M * NLVL * 2;
    float* tmp     = valbuf;

    __bf16* bb = (__bf16*)ws;
    __bf16* qb16    = bb; bb += M * DMODEL;   // q; reused as sampled
    __bf16* curb16  = bb; bb += M * DMODEL;   // layer input (bf16)
    __bf16* srcAb16 = bb; bb += M * DMODEL;
    __bf16* ffb16   = bb; bb += M * DFF;
    __bf16* Woffb   = bb; bb += (size_t)NLAYERS * DMODEL * 256;
    __bf16* Wattnb  = bb; bb += (size_t)NLAYERS * DMODEL * 128;
    __bf16* Wvalb   = bb; bb += (size_t)NLAYERS * DMODEL * DMODEL;
    __bf16* Woutb   = bb; bb += (size_t)NLAYERS * DMODEL * DMODEL;
    __bf16* Wff1b   = bb; bb += (size_t)NLAYERS * DMODEL * DFF;
    __bf16* Wff2b   = bb; bb += (size_t)NLAYERS * DFF * DMODEL;

    const dim3 blk(256);
    auto cvt = [&](const float* in, __bf16* out, size_t n) {
        int n4 = (int)(n / 4);
        cvt_bf16_kernel<<<(n4 + 255) / 256, blk, 0, stream>>>(in, out, n4);
    };

    // ---- one-time (per launch) conversions ----
    cvt(W_off,  Woffb,  (size_t)NLAYERS * DMODEL * 256);
    cvt(W_attn, Wattnb, (size_t)NLAYERS * DMODEL * 128);
    cvt(W_val,  Wvalb,  (size_t)NLAYERS * DMODEL * DMODEL);
    cvt(W_out,  Woutb,  (size_t)NLAYERS * DMODEL * DMODEL);
    cvt(W_ff1,  Wff1b,  (size_t)NLAYERS * DMODEL * DFF);
    cvt(W_ff2,  Wff2b,  (size_t)NLAYERS * DFF * DMODEL);
    cvt(src,    curb16, M * DMODEL);

    refpoints_kernel<<<(BSZ * SEQ + 255) / 256, blk, 0, stream>>>(vr, refbuf);

    auto gemm_f32 = [&](const __bf16* Ab, const __bf16* Wb, const float* bbb,
                        float* Cm, int K, int N) {
        dim3 grid(N / 128, (unsigned)((M + 127) / 128));
        gemm_bias_kernel<false, true, false><<<grid, blk, 0, stream>>>(
            Ab, Wb, bbb, Cm, (__bf16*)nullptr, (int)M, K, N);
    };
    auto gemm_relu_b16 = [&](const __bf16* Ab, const __bf16* Wb, const float* bbb,
                             __bf16* Cm, int K, int N) {
        dim3 grid(N / 128, (unsigned)((M + 127) / 128));
        gemm_bias_kernel<true, false, true><<<grid, blk, 0, stream>>>(
            Ab, Wb, bbb, (float*)nullptr, Cm, (int)M, K, N);
    };

    const float* cur = src;  // fp32 layer input (residual path)
    for (int l = 0; l < NLAYERS; ++l) {
        const int n4 = (int)(M * DMODEL / 4);
        addq_bf16_kernel<<<(n4 + 255) / 256, blk, 0, stream>>>(
            (const float4*)cur, (const float4*)pos, qb16, n4);

        gemm_f32(curb16, Wvalb  + (size_t)l * DMODEL * DMODEL, b_val  + (size_t)l * DMODEL, valbuf,  DMODEL, DMODEL);
        gemm_f32(qb16,   Woffb  + (size_t)l * DMODEL * 256,    b_off  + (size_t)l * 256,    offbuf,  DMODEL, 256);
        gemm_f32(qb16,   Wattnb + (size_t)l * DMODEL * 128,    b_attn + (size_t)l * 128,    attnbuf, DMODEL, 128);

        softmax16_kernel<<<(int)((M * NHEAD + 255) / 256), blk, 0, stream>>>(
            attnbuf, (int)(M * NHEAD));

        deform_kernel<<<(int)(M * NHEAD / 8), blk, 0, stream>>>(
            valbuf, offbuf, attnbuf, refbuf, qb16 /* sampled, bf16 */);

        gemm_f32(qb16, Woutb + (size_t)l * DMODEL * DMODEL, b_out + (size_t)l * DMODEL, tmp, DMODEL, DMODEL);

        resid_ln_kernel<true><<<(int)((M + 7) / 8), blk, 0, stream>>>(
            cur, tmp, g1 + (size_t)l * DMODEL, be1 + (size_t)l * DMODEL,
            srcA, srcAb16, (int)M);

        gemm_relu_b16(srcAb16, Wff1b + (size_t)l * DMODEL * DFF, b_ff1 + (size_t)l * DFF, ffb16, DMODEL, DFF);
        gemm_f32(ffb16, Wff2b + (size_t)l * DFF * DMODEL, b_ff2 + (size_t)l * DMODEL, tmp, DFF, DMODEL);

        float* outp = (l == NLAYERS - 1) ? (float*)d_out : srcB;
        resid_ln_kernel<true><<<(int)((M + 7) / 8), blk, 0, stream>>>(
            srcA, tmp, g2 + (size_t)l * DMODEL, be2 + (size_t)l * DMODEL,
            outp, curb16, (int)M);
        cur = srcB;
    }
}
